// MOE_72181220376978
// MI455X (gfx1250) — matrix operand order repaired and checked by reading, not compiled, not run on hardware
//
#include <hip/hip_runtime.h>
#include <hip/hip_bf16.h>
#include <stdint.h>

// ---------------------------------------------------------------------------
// KAN-MoE on MI455X (gfx1250).
//
// kan_linear(x) = silu(x) @ Wb^T + einsum(bsplines(x), Ws)
// folds into ONE GEMM on augmented features: per input feature i, the 9
// values [silu(x_i), B_0..B_7(x_i)] against [Wb | Ws] packed column-blocks.
//
// Pipeline (all bf16 GEMMs on v_wmma_f32_16x16x32_bf16, f32 accum):
//   F0 = featurize(x)              (8192 x 4608) bf16   -- fits in 192MB L2
//   logits = F0 @ WrT (4608x16), softmax -> probs (8192 x 8)
//   per expert n (sequential):
//     H  = F0 @ W1T[n]  (4608x128)         -> f32
//     F1 = featurize(H) (8192 x 1152) bf16
//     out += probs[:,n] * (F1 @ W2T[n] (1152x512))
//
// ~155 GFLOP matrix work, ~100MB unique bytes => WMMA-bound, not HBM-bound.
// B tiles are DMA'd to LDS with GLOBAL_LOAD_ASYNC_TO_LDS_B128 (ASYNCcnt),
// double-buffered so the copy overlaps the WMMAs.
// ---------------------------------------------------------------------------

typedef __attribute__((ext_vector_type(16))) __bf16 v16bf;
typedef __attribute__((ext_vector_type(8)))  __bf16 v8bf;
typedef __attribute__((ext_vector_type(8)))  float  v8f;

union V16U { v16bf v; v8bf h[2]; };

#define T_TOK   8192
#define D_IN    512
#define HID     128
#define NEXP    8
#define KT0     (D_IN * 9)   // 4608
#define KT1     (HID * 9)    // 1152

static __device__ __forceinline__ unsigned short f2bf(float f) {
    unsigned int u = __float_as_uint(f);
    unsigned int r = u + 0x7FFFu + ((u >> 16) & 1u);   // round-to-nearest-even
    return (unsigned short)(r >> 16);
}

static __device__ __forceinline__ float silu(float x) {
    return x * (1.0f / (1.0f + __expf(-x)));
}

// Async DMA: 16 bytes global -> LDS per lane, tracked by ASYNCcnt.
static __device__ __forceinline__ void async_b128_to_lds(const void* gsrc, void* lds_dst) {
    unsigned lds_off = (unsigned)(size_t)lds_dst;              // low 32b = LDS byte addr
    unsigned long long ga = (unsigned long long)(size_t)gsrc;
    asm volatile("global_load_async_to_lds_b128 %0, %1, off"
                 :: "v"(lds_off), "v"(ga) : "memory");
}
static __device__ __forceinline__ void wait_async0() {
    asm volatile("s_wait_asynccnt 0x0" ::: "memory");
}

// Cox-de Boor, uniform knots t_j = -1 + (j-3)*0.4, order 3 -> 8 bases.
static __device__ __forceinline__ void bspline8(float x, float* __restrict__ b) {
    float kn[12];
#pragma unroll
    for (int j = 0; j < 12; ++j) kn[j] = -1.0f + (float)(j - 3) * 0.4f;
    float t[11];
#pragma unroll
    for (int j = 0; j < 11; ++j) t[j] = (x >= kn[j] && x < kn[j + 1]) ? 1.0f : 0.0f;
#pragma unroll
    for (int k = 1; k <= 3; ++k) {
        float inv = 1.0f / ((float)k * 0.4f);
        for (int j = 0; j < 11 - k; ++j)
            t[j] = ((x - kn[j]) * t[j] + (kn[j + k + 1] - x) * t[j + 1]) * inv;
    }
#pragma unroll
    for (int j = 0; j < 8; ++j) b[j] = t[j];
}

// ---------------- featurize: X (T x Din f32) -> F (T x Din*9 bf16) ---------
__global__ __launch_bounds__(256) void kan_featurize(const float* __restrict__ X,
                                                     unsigned short* __restrict__ F,
                                                     int Din) {
    int idx = blockIdx.x * 256 + threadIdx.x;
    int t = idx / Din, i = idx % Din;
    float xv = X[(size_t)t * Din + i];
    float bas[8];
    bspline8(xv, bas);
    unsigned short* row = F + (size_t)t * (Din * 9) + (size_t)i * 9;
    row[0] = f2bf(silu(xv));
#pragma unroll
    for (int k = 0; k < 8; ++k) row[1 + k] = f2bf(bas[k]);
}

// --------------- weight packing into transposed bf16 (Kt x Nout) -----------
// Router: (4608 x 16), cols 8..15 zero-padded.
__global__ __launch_bounds__(256) void pack_router(const float* __restrict__ Wb,
                                                   const float* __restrict__ Ws,
                                                   unsigned short* __restrict__ Wt) {
    int idx = blockIdx.x * 256 + threadIdx.x;           // 4608*16
    int r = idx >> 4, o = idx & 15;
    int i = r / 9, j = r % 9;
    float v = 0.0f;
    if (o < 8) v = (j == 0) ? Wb[(size_t)o * D_IN + i]
                            : Ws[((size_t)o * D_IN + i) * 8 + (j - 1)];
    Wt[(size_t)r * 16 + o] = f2bf(v);
}

// Expert layer 1: per n, (4608 x 128)
__global__ __launch_bounds__(256) void pack_w1(const float* __restrict__ Wb,
                                               const float* __restrict__ Ws,
                                               unsigned short* __restrict__ Wt) {
    int idx = blockIdx.x * 256 + threadIdx.x;           // 8*4608*128
    int o = idx & (HID - 1);
    int r = (idx >> 7) % KT0;
    int n = idx / (KT0 * HID);
    int i = r / 9, j = r % 9;
    size_t wo = (size_t)n * HID + o;
    float v = (j == 0) ? Wb[wo * D_IN + i] : Ws[(wo * D_IN + i) * 8 + (j - 1)];
    Wt[((size_t)n * KT0 + r) * HID + o] = f2bf(v);
}

// Expert layer 2: per n, (1152 x 512)
__global__ __launch_bounds__(256) void pack_w2(const float* __restrict__ Wb,
                                               const float* __restrict__ Ws,
                                               unsigned short* __restrict__ Wt) {
    int idx = blockIdx.x * 256 + threadIdx.x;           // 8*1152*512
    int o = idx & (D_IN - 1);
    int r = (idx >> 9) % KT1;
    int n = idx / (KT1 * D_IN);
    int i = r / 9, j = r % 9;
    size_t wo = (size_t)n * D_IN + o;
    float v = (j == 0) ? Wb[wo * HID + i] : Ws[(wo * HID + i) * 8 + (j - 1)];
    Wt[((size_t)n * KT1 + r) * D_IN + o] = f2bf(v);
}

// --------------- router GEMM: one wave per 16x16 tile, Nout=16 -------------
__global__ __launch_bounds__(32) void router_gemm(const unsigned short* __restrict__ A,
                                                  const unsigned short* __restrict__ Wr,
                                                  float* __restrict__ logits) {
    const int lane = threadIdx.x;
    const int m0 = blockIdx.x * 16;
    const int mrow = lane & 15, khalf = (lane >> 4) * 8;
    v8f acc = {};
    for (int k0 = 0; k0 < KT0; k0 += 32) {
        const unsigned short* ar = A + (size_t)(m0 + mrow) * KT0 + k0 + khalf;
        V16U a; a.h[0] = *(const v8bf*)ar; a.h[1] = *(const v8bf*)(ar + 16);
        const unsigned short* br = Wr + (size_t)(k0 + khalf) * 16 + mrow;
        V16U b;
#pragma unroll
        for (int e = 0; e < 16; ++e)
            b.v[e] = __builtin_bit_cast(__bf16, br[(e < 8 ? e : e + 8) * 16]);
        acc = __builtin_amdgcn_wmma_f32_16x16x32_bf16(false, a.v, false, b.v,
                                                      (short)0, acc, false, false);
    }
#pragma unroll
    for (int r = 0; r < 8; ++r)
        logits[(size_t)(m0 + r + 8 * (lane >> 4)) * 16 + (lane & 15)] = acc[r];
}

// --------------- softmax over 8 experts ------------------------------------
__global__ __launch_bounds__(256) void router_softmax(const float* __restrict__ logits,
                                                      float* __restrict__ probs) {
    int t = blockIdx.x * 256 + threadIdx.x;
    float l[8], mx = -1e30f;
#pragma unroll
    for (int j = 0; j < 8; ++j) { l[j] = logits[(size_t)t * 16 + j]; mx = fmaxf(mx, l[j]); }
    float s = 0.0f;
#pragma unroll
    for (int j = 0; j < 8; ++j) { l[j] = __expf(l[j] - mx); s += l[j]; }
    float inv = 1.0f / s;
#pragma unroll
    for (int j = 0; j < 8; ++j) probs[(size_t)t * 8 + j] = l[j] * inv;
}

// --------------- main tiled GEMM -------------------------------------------
// 8 waves/WG; WG tile 128(M) x 64(N); B-tile (32x64 bf16) in LDS, double-
// buffered via async global->LDS DMA (ASYNCcnt) so copy overlaps WMMA.
// MODE 0: C[m*Nout+n] = acc              (layer-1, f32 hidden)
// MODE 1: C[m*512+n] += probs[m*8+e]*acc (layer-2, fused routing)
template <int MODE>
__global__ __launch_bounds__(256) void kan_gemm(const unsigned short* __restrict__ A,
                                                const unsigned short* __restrict__ B,
                                                int Kt, int Nout,
                                                float* __restrict__ C,
                                                const float* __restrict__ probs,
                                                int expert) {
    __shared__ __align__(16) unsigned short lds[2][32 * 64];
    const int tid  = threadIdx.x;
    const int wave = tid >> 5, lane = tid & 31;
    const int gm = blockIdx.x * 128, gn = blockIdx.y * 64;
    const int krow = tid >> 3, ncol8 = (tid & 7) * 8;   // this thread's 16B chunk
    const int mrow = lane & 15, khalf = (lane >> 4) * 8;
    const int nk = Kt / 32;

    const unsigned short* bsrc = B + (size_t)krow * Nout + gn + ncol8;
    void* ldst[2] = { &lds[0][krow * 64 + ncol8], &lds[1][krow * 64 + ncol8] };

    // preload B tile 0 via async DMA
    async_b128_to_lds(bsrc, ldst[0]);
    wait_async0();
    __syncthreads();

    v8f acc[4];
#pragma unroll
    for (int s = 0; s < 4; ++s) acc[s] = (v8f){};

    for (int kt = 0; kt < nk; ++kt) {
        const int cur = kt & 1;
        if (kt + 1 < nk)                                   // DMA next tile
            async_b128_to_lds(bsrc + (size_t)(kt + 1) * 32 * Nout, ldst[cur ^ 1]);

        const unsigned short* ar =
            A + (size_t)(gm + wave * 16 + mrow) * Kt + kt * 32 + khalf;
        if (kt + 1 < nk) __builtin_prefetch(ar + 32, 0, 1);   // global_prefetch_b8
        V16U a; a.h[0] = *(const v8bf*)ar; a.h[1] = *(const v8bf*)(ar + 16);

#pragma unroll
        for (int s = 0; s < 4; ++s) {
            v16bf b;
#pragma unroll
            for (int e = 0; e < 16; ++e)
                b[e] = __builtin_bit_cast(__bf16, lds[cur][(khalf + (e < 8 ? e : e + 8)) * 64 + s * 16 + mrow]);  // lane%16 = N col, elements over K
            acc[s] = __builtin_amdgcn_wmma_f32_16x16x32_bf16(false, a.v, false, b,
                                                             (short)0, acc[s],
                                                             false, false);
        }
        wait_async0();          // own DMA done ...
        __syncthreads();        // ... and every wave's DMA done
    }

#pragma unroll
    for (int s = 0; s < 4; ++s) {
        int n = gn + s * 16 + (lane & 15);
#pragma unroll
        for (int r = 0; r < 8; ++r) {
            int m = gm + wave * 16 + r + 8 * (lane >> 4);
            if (MODE == 0) {
                C[(size_t)m * Nout + n] = acc[s][r];
            } else {
                float p = probs[(size_t)m * 8 + expert];
                C[(size_t)m * Nout + n] += p * acc[s][r];
            }
        }
    }
}

__global__ __launch_bounds__(256) void zero_f32(float* __restrict__ p, int n) {
    int i = blockIdx.x * 256 + threadIdx.x;
    if (i < n) p[i] = 0.0f;
}

// ---------------------------------------------------------------------------
extern "C" void kernel_launch(void* const* d_in, const int* in_sizes, int n_in,
                              void* d_out, int out_size, void* d_ws, size_t ws_size,
                              hipStream_t stream) {
    const float* x   = (const float*)d_in[0];
    const float* rb  = (const float*)d_in[1];
    const float* rs  = (const float*)d_in[2];
    const float* e1b = (const float*)d_in[3];
    const float* e1s = (const float*)d_in[4];
    const float* e2b = (const float*)d_in[5];
    const float* e2s = (const float*)d_in[6];
    float* out = (float*)d_out;

    // workspace layout (bytes)
    char* ws = (char*)d_ws;
    unsigned short* F0  = (unsigned short*)(ws);                     // 8192*4608*2 = 75,497,472
    unsigned short* WrT = (unsigned short*)(ws + 75497472);          // 4608*16*2   =    147,456
    unsigned short* W1T = (unsigned short*)(ws + 75644928);          // 8*4608*128*2=  9,437,184
    unsigned short* W2T = (unsigned short*)(ws + 85082112);          // 8*1152*512*2=  9,437,184
    float*          H   = (float*)(ws + 94519296);                   // 8192*128*4  =  4,194,304
    unsigned short* F1  = (unsigned short*)(ws + 98713600);          // 8192*1152*2 = 18,874,368
    float*          LG  = (float*)(ws + 117587968);                  // 8192*16*4   =    524,288
    float*          PR  = (float*)(ws + 118112256);                  // 8192*8*4    =    262,144

    zero_f32<<<(T_TOK * D_IN + 255) / 256, 256, 0, stream>>>(out, T_TOK * D_IN);

    // featurize inputs + pack augmented weights (bf16, transposed to Kt x Nout)
    kan_featurize<<<T_TOK * D_IN / 256, 256, 0, stream>>>(x, F0, D_IN);
    pack_router<<<KT0 * 16 / 256, 256, 0, stream>>>(rb, rs, WrT);
    pack_w1<<<NEXP * KT0 * HID / 256, 256, 0, stream>>>(e1b, e1s, W1T);
    pack_w2<<<NEXP * KT1 * D_IN / 256, 256, 0, stream>>>(e2b, e2s, W2T);

    // router
    router_gemm<<<T_TOK / 16, 32, 0, stream>>>(F0, WrT, LG);
    router_softmax<<<T_TOK / 256, 256, 0, stream>>>(LG, PR);

    // experts (sequential on stream; F0 stays resident in the 192MB L2)
    for (int n = 0; n < NEXP; ++n) {
        dim3 g1(T_TOK / 128, HID / 64);
        kan_gemm<0><<<g1, 256, 0, stream>>>(F0, W1T + (size_t)n * KT0 * HID,
                                            KT0, HID, H, nullptr, 0);
        kan_featurize<<<T_TOK * HID / 256, 256, 0, stream>>>(H, F1, HID);
        dim3 g2(T_TOK / 128, D_IN / 64);
        kan_gemm<1><<<g2, 256, 0, stream>>>(F1, W2T + (size_t)n * KT1 * D_IN,
                                            KT1, D_IN, out, PR, n);
    }
}
